// CQEncoder_54236847014294
// MI455X (gfx1250) — matrix operand order repaired and checked by reading, not compiled, not run on hardware
//
#include <hip/hip_runtime.h>
#include <hip/hip_bf16.h>
#include <math.h>

typedef __bf16 bf16;
typedef __attribute__((ext_vector_type(16))) __bf16 v16bf;
typedef __attribute__((ext_vector_type(8)))  float  v8f;

#define B_    64
#define NC    1024
#define NQ    128
#define Fdim  512
#define Hh    8
#define DHd   64
#define NHIDd 512
#define ALPHAc 0.2f
#define NEGV  (-9e15f)

#define WMMA_BF16(a, b, c) \
  __builtin_amdgcn_wmma_f32_16x16x32_bf16(false, (a), false, (b), (short)0, (c), false, false)

// ---------------- WMMA fragment loaders (wave32, 16x16x32 bf16) ----------------
// A (16xK tile, row-major, 16-bit): lane m=L%16, base=(L<16)?0:8;
// halves 0..7 = A[m][k0+base .. +7], halves 8..15 = A[m][k0+16+base .. +7]
__device__ __forceinline__ v16bf load_a_frag(const bf16* __restrict__ A, int lda) {
  int lane = threadIdx.x & 31;
  int m  = lane & 15;
  int kb = (lane >> 4) << 3; // 0 or 8
  const bf16* p = A + (size_t)m * lda + kb;
  v16bf f;
#pragma unroll
  for (int j = 0; j < 8; ++j) { f[j] = p[j]; f[8 + j] = p[16 + j]; }
  return f;
}
// B via B^T stored [N][K] row-major: lane n=L%16, base=(L<16)?0:8; halves 0..7 = k base..+7, 8..15 = k 16+base..+7
__device__ __forceinline__ v16bf load_bt_frag(const bf16* __restrict__ Bt, int ldbt) {
  int lane = threadIdx.x & 31;
  int n  = lane & 15;
  int kb = (lane >> 4) << 3; // 0 or 8
  const bf16* p = Bt + (size_t)n * ldbt + kb;
  v16bf f;
#pragma unroll
  for (int j = 0; j < 8; ++j) { f[j] = p[j]; f[8 + j] = p[16 + j]; }
  return f;
}

// ---------------- elementwise / prep kernels ----------------
__global__ void k_cvt_bf16(const float* __restrict__ x, bf16* __restrict__ y, long n) {
  long i = (long)blockIdx.x * blockDim.x + threadIdx.x;
  if (i < n) y[i] = (bf16)x[i];
}

// WcatT[(h*DH+d)][f] = W[h][f][d]   (W pre-offset to layer; [H,F,DH])
__global__ void k_prep_wcat(const float* __restrict__ W, bf16* __restrict__ WcatT) {
  int i = blockIdx.x * blockDim.x + threadIdx.x; // NHID*F
  if (i >= NHIDd * Fdim) return;
  int c = i / Fdim, f = i % Fdim;
  int h = c / DHd, d = c % DHd;
  WcatT[i] = (bf16)W[((size_t)h * Fdim + f) * DHd + d];
}

// wa1[h][f] = sum_d W[h][f][d] * a1[h][d]
__global__ void k_prep_wa1(const float* __restrict__ W, const float* __restrict__ avec,
                           float* __restrict__ wa1) {
  int i = blockIdx.x * blockDim.x + threadIdx.x; // H*F
  if (i >= Hh * Fdim) return;
  int h = i / Fdim, f = i % Fdim;
  const float* wp = W + ((size_t)h * Fdim + f) * DHd;
  const float* ap = avec + h * 2 * DHd;
  float s = 0.f;
#pragma unroll 8
  for (int d = 0; d < DHd; ++d) s += wp[d] * ap[d];
  wa1[i] = s;
}

// pT[i][j][f] = p[i][f][j]   (p pre-offset to layer; [4,F,NHID])
__global__ void k_prep_pT(const float* __restrict__ p, bf16* __restrict__ pT) {
  long i = (long)blockIdx.x * blockDim.x + threadIdx.x; // 4*NHID*F
  if (i >= 4L * NHIDd * Fdim) return;
  long mi = i / ((long)NHIDd * Fdim);
  long r  = i % ((long)NHIDd * Fdim);
  long j = r / Fdim, f = r % Fdim;
  pT[i] = (bf16)p[mi * Fdim * NHIDd + f * NHIDd + j];
}

// ---------------- projection GEMM: proj = x_b @ Wcat, scattered to projT[bh][d][m] ----
// 32 (rows) x 64 (cols) per wave, single-buffered: 6 b128-load clause -> 8 WMMAs per k-step
__global__ void k_proj(const bf16* __restrict__ A, const bf16* __restrict__ WcatT,
                       bf16* __restrict__ projT, int M) {
  int row0 = blockIdx.x * 32;
  int col0 = blockIdx.y * 64;
  const bf16* pA0 = A + (size_t)row0 * Fdim;
  const bf16* pA1 = A + (size_t)(row0 + 16) * Fdim;
  v8f acc[2][4] = {};
  for (int k0 = 0; k0 < Fdim; k0 += 32) {
    v16bf a0 = load_a_frag(pA0 + k0, Fdim);
    v16bf a1 = load_a_frag(pA1 + k0, Fdim);
    v16bf bfr[4];
#pragma unroll
    for (int t = 0; t < 4; ++t)
      bfr[t] = load_bt_frag(WcatT + (size_t)(col0 + 16 * t) * Fdim + k0, Fdim);
#pragma unroll
    for (int t = 0; t < 4; ++t) {
      acc[0][t] = WMMA_BF16(a0, bfr[t], acc[0][t]);
      acc[1][t] = WMMA_BF16(a1, bfr[t], acc[1][t]);
    }
  }
  int lane = threadIdx.x;
  int rhi = (lane >> 4) << 3, nn = lane & 15;
#pragma unroll
  for (int rg = 0; rg < 2; ++rg)
#pragma unroll
    for (int t = 0; t < 4; ++t)
#pragma unroll
      for (int v = 0; v < 8; ++v) {
        int r = row0 + 16 * rg + rhi + v;
        int c = col0 + 16 * t + nn;
        int b = r / M, m = r % M;
        int h = c >> 6, d = c & 63;
        projT[(((size_t)b * Hh + h) * DHd + d) * M + m] = (bf16)acc[rg][t][v];
      }
}

// ea[(b*H+h)*N + n] = x_a[b,n,:] . wa1[h,:]
__global__ void k_ea(const float* __restrict__ x, const float* __restrict__ wa1,
                     float* __restrict__ ea, int N) {
  long t = (long)blockIdx.x * blockDim.x + threadIdx.x;
  long total = (long)B_ * N * Hh;
  if (t >= total) return;
  int h = (int)(t & 7);
  long row = t >> 3; // b*N + n
  int b = (int)(row / N), n = (int)(row % N);
  const float* xp = x + row * Fdim;
  const float* wp = wa1 + h * Fdim;
  float s = 0.f;
#pragma unroll 8
  for (int f = 0; f < Fdim; ++f) s += xp[f] * wp[f];
  ea[((size_t)b * Hh + h) * N + n] = s;
}

// eb[bh*M+m] = sum_d projT[bh][d][m] * a2[h][d]
__global__ void k_eb(const bf16* __restrict__ projT, const float* __restrict__ avec,
                     float* __restrict__ eb, int M) {
  long t = (long)blockIdx.x * blockDim.x + threadIdx.x;
  long total = (long)B_ * Hh * M;
  if (t >= total) return;
  int m = (int)(t % M);
  long bh = t / M;
  int h = (int)(bh % Hh);
  const float* ap = avec + h * 2 * DHd + DHd;
  const bf16* pp = projT + (size_t)bh * DHd * M + m;
  float s = 0.f;
#pragma unroll 8
  for (int d = 0; d < DHd; ++d) s += (float)pp[(size_t)d * M] * ap[d];
  eb[t] = s;
}

// online softmax over N per (b,h,m) column
__global__ void k_smax_stats(const float* __restrict__ ea, const float* __restrict__ eb,
                             const int* __restrict__ adj, float* __restrict__ colmax,
                             float* __restrict__ colrcp, int N, int M) {
  long t = (long)blockIdx.x * blockDim.x + threadIdx.x;
  long total = (long)B_ * Hh * M;
  if (t >= total) return;
  int m = (int)(t % M);
  long bh = t / M;
  int b = (int)(bh / Hh);
  float ebv = eb[t];
  const float* eap = ea + bh * N;
  const int* ap = adj + (size_t)b * N * M + m;
  float mx = -INFINITY, sum = 0.f;
  for (int n = 0; n < N; ++n) {
    float e = eap[n] + ebv;
    e = e > 0.f ? e : ALPHAc * e;
    if (ap[(size_t)n * M] == 0) e = NEGV;
    if (e > mx) { sum = sum * __expf(mx - e) + 1.f; mx = e; }
    else        { sum += __expf(e - mx); }
  }
  colmax[t] = mx;
  colrcp[t] = 1.f / sum;
}

// materialize att (bf16, row-major [bh][n][m])
__global__ void k_att(const float* __restrict__ ea, const float* __restrict__ eb,
                      const int* __restrict__ adj, const float* __restrict__ colmax,
                      const float* __restrict__ colrcp, bf16* __restrict__ att,
                      int N, int M) {
  long t = (long)blockIdx.x * blockDim.x + threadIdx.x;
  long total = (long)B_ * Hh * N * M;
  if (t >= total) return;
  int m = (int)(t % M);
  long r = t / M;
  int n = (int)(r % N);
  long bh = r / N;
  int b = (int)(bh / Hh);
  float e = ea[bh * N + n] + eb[bh * M + m];
  e = e > 0.f ? e : ALPHAc * e;
  if (adj[(size_t)b * N * M + (size_t)n * M + m] == 0) e = NEGV;
  float w = __expf(e - colmax[bh * M + m]) * colrcp[bh * M + m];
  att[t] = (bf16)w;
}

// h = elu(att @ Wb) -> agg[b,n, h*64+d] (bf16); 32x64 tile, single-buffered; K=M
__global__ void k_hgemm(const bf16* __restrict__ att, const bf16* __restrict__ projT,
                        bf16* __restrict__ agg, int N, int M) {
  int row0 = blockIdx.x * 32;
  long bh = blockIdx.y;
  int b = (int)(bh / Hh), h = (int)(bh % Hh);
  const bf16* A  = att   + (size_t)bh * N * M;
  const bf16* Bt = projT + (size_t)bh * DHd * M;
  const bf16* pA0 = A + (size_t)row0 * M;
  const bf16* pA1 = A + (size_t)(row0 + 16) * M;
  v8f acc[2][4] = {};
  for (int k0 = 0; k0 < M; k0 += 32) {
    v16bf a0 = load_a_frag(pA0 + k0, M);
    v16bf a1 = load_a_frag(pA1 + k0, M);
    v16bf bfr[4];
#pragma unroll
    for (int t = 0; t < 4; ++t)
      bfr[t] = load_bt_frag(Bt + (size_t)(16 * t) * M + k0, M);
#pragma unroll
    for (int t = 0; t < 4; ++t) {
      acc[0][t] = WMMA_BF16(a0, bfr[t], acc[0][t]);
      acc[1][t] = WMMA_BF16(a1, bfr[t], acc[1][t]);
    }
  }
  int lane = threadIdx.x;
  int rhi = (lane >> 4) << 3, nn = lane & 15;
#pragma unroll
  for (int rg = 0; rg < 2; ++rg)
#pragma unroll
    for (int t = 0; t < 4; ++t)
#pragma unroll
      for (int v = 0; v < 8; ++v) {
        int n = row0 + 16 * rg + rhi + v;
        int d = 16 * t + nn;
        float x = acc[rg][t][v];
        x = x > 0.f ? x : (__expf(x) - 1.f);
        agg[((size_t)b * N + n) * NHIDd + h * DHd + d] = (bf16)x;
      }
}

// fusion: new_a = a@p0 + g@p1; f = sigmoid(a@p2 + g@p3); out = f*new_a + (1-f)*a
// 32 rows x 16 cols per wave, single-buffered: 8 loads -> 8 WMMAs per k-step
__global__ void k_fusion(const bf16* __restrict__ aB, const bf16* __restrict__ gB,
                         const bf16* __restrict__ pT, const float* __restrict__ aF,
                         float* __restrict__ outF, bf16* __restrict__ outB) {
  int row0 = blockIdx.x * 32;
  int col0 = blockIdx.y * 16;
  const bf16* p0 = pT + (size_t)col0 * Fdim;
  const bf16* p1 = pT + 1ull * NHIDd * Fdim + (size_t)col0 * Fdim;
  const bf16* p2 = pT + 2ull * NHIDd * Fdim + (size_t)col0 * Fdim;
  const bf16* p3 = pT + 3ull * NHIDd * Fdim + (size_t)col0 * Fdim;
  const bf16* pa0 = aB + (size_t)row0 * Fdim;
  const bf16* pa1 = aB + (size_t)(row0 + 16) * Fdim;
  const bf16* pg0 = gB + (size_t)row0 * Fdim;
  const bf16* pg1 = gB + (size_t)(row0 + 16) * Fdim;
  v8f acc0[2] = {}, acc1[2] = {};
  for (int k0 = 0; k0 < Fdim; k0 += 32) {
    v16bf fa[2], fg[2], w[4];
    fa[0] = load_a_frag(pa0 + k0, Fdim); fa[1] = load_a_frag(pa1 + k0, Fdim);
    fg[0] = load_a_frag(pg0 + k0, Fdim); fg[1] = load_a_frag(pg1 + k0, Fdim);
    w[0] = load_bt_frag(p0 + k0, Fdim);  w[1] = load_bt_frag(p1 + k0, Fdim);
    w[2] = load_bt_frag(p2 + k0, Fdim);  w[3] = load_bt_frag(p3 + k0, Fdim);
#pragma unroll
    for (int rg = 0; rg < 2; ++rg) {
      acc0[rg] = WMMA_BF16(fa[rg], w[0], acc0[rg]);
      acc0[rg] = WMMA_BF16(fg[rg], w[1], acc0[rg]);
      acc1[rg] = WMMA_BF16(fa[rg], w[2], acc1[rg]);
      acc1[rg] = WMMA_BF16(fg[rg], w[3], acc1[rg]);
    }
  }
  int lane = threadIdx.x;
  int rhi = (lane >> 4) << 3, nn = lane & 15;
#pragma unroll
  for (int rg = 0; rg < 2; ++rg)
#pragma unroll
    for (int v = 0; v < 8; ++v) {
      long r = row0 + 16 * rg + rhi + v;
      int c = col0 + nn;
      float f  = 1.f / (1.f + __expf(-acc1[rg][v]));
      float av = aF[r * NHIDd + c];
      float o  = f * acc0[rg][v] + (1.f - f) * av;
      outF[r * NHIDd + c] = o;
      outB[r * NHIDd + c] = (bf16)o;
    }
}

// ---------------- host orchestration ----------------
extern "C" void kernel_launch(void* const* d_in, const int* in_sizes, int n_in,
                              void* d_out, int out_size, void* d_ws, size_t ws_size,
                              hipStream_t stream) {
  (void)in_sizes; (void)n_in; (void)out_size; (void)ws_size;
  const float* clip_in  = (const float*)d_in[0];
  const float* q_in     = (const float*)d_in[1];
  const int*   adj_q2c  = (const int*)d_in[2]; // clip2questionadj [B,NC,NQ] (agg question->clip)
  const int*   adj_c2q  = (const int*)d_in[3]; // question2clipadj [B,NQ,NC] (agg clip->question)
  const float* W1 = (const float*)d_in[4];  const float* a1 = (const float*)d_in[5];
  const float* W2 = (const float*)d_in[6];  const float* a2 = (const float*)d_in[7];
  const float* f1 = (const float*)d_in[8];  const float* f2 = (const float*)d_in[9];

  const long nClip = (long)B_ * NC * NHIDd;   // 33554432
  const long nQ    = (long)B_ * NQ * NHIDd;   // 4194304

  char* w = (char*)d_ws;
  auto alloc = [&](size_t bytes) -> char* {
    char* p = w; w += (bytes + 255) & ~(size_t)255; return p;
  };
  bf16*  clipB[2] = { (bf16*)alloc(nClip * 2), (bf16*)alloc(nClip * 2) };
  bf16*  qB[2]    = { (bf16*)alloc(nQ * 2),    (bf16*)alloc(nQ * 2)    };
  float* clipF1   = (float*)alloc(nClip * 4);
  float* qF1      = (float*)alloc(nQ * 4);
  bf16*  projT    = (bf16*)alloc((size_t)B_ * Hh * NC * DHd * 2);
  bf16*  attW     = (bf16*)alloc((size_t)B_ * Hh * NC * NQ * 2);
  bf16*  aggB     = (bf16*)alloc((size_t)B_ * NC * NHIDd * 2);
  float* eaBuf    = (float*)alloc((size_t)B_ * Hh * NC * 4);
  float* ebBuf    = (float*)alloc((size_t)B_ * Hh * NC * 4);
  float* colmax   = (float*)alloc((size_t)B_ * Hh * NC * 4);
  float* colrcp   = (float*)alloc((size_t)B_ * Hh * NC * 4);
  float* wa1      = (float*)alloc((size_t)Hh * Fdim * 4);
  bf16*  WcatT    = (bf16*)alloc((size_t)NHIDd * Fdim * 2);
  bf16*  pT       = (bf16*)alloc(4ull * NHIDd * Fdim * 2);

  k_cvt_bf16<<<(unsigned)((nClip + 255) / 256), 256, 0, stream>>>(clip_in, clipB[0], nClip);
  k_cvt_bf16<<<(unsigned)((nQ + 255) / 256), 256, 0, stream>>>(q_in, qB[0], nQ);

  const float* clipF_cur = clip_in;
  const float* qF_cur    = q_in;
  float* clipOut = (float*)d_out;
  float* qOut    = (float*)d_out + nClip;
  int cur = 0;

  for (int l = 0; l < 2; ++l) {
    int nxt = cur ^ 1;
    float* clipF_next = (l == 0) ? clipF1 : clipOut;
    float* qF_next    = (l == 0) ? qF1    : qOut;

    // ===== direction 1: aggregate question into clip (N=NC, M=NQ) =====
    {
      const float* W  = W1 + (size_t)l * Hh * Fdim * DHd;
      const float* av = a1 + (size_t)l * Hh * 2 * DHd;
      const float* fp = f1 + (size_t)l * 4 * Fdim * NHIDd;
      const int N = NC, M = NQ;
      k_prep_wcat<<<(NHIDd * Fdim + 255) / 256, 256, 0, stream>>>(W, WcatT);
      k_prep_wa1<<<(Hh * Fdim + 255) / 256, 256, 0, stream>>>(W, av, wa1);
      k_prep_pT<<<(4 * NHIDd * Fdim + 255) / 256, 256, 0, stream>>>(fp, pT);
      dim3 gp(B_ * M / 32, NHIDd / 64);
      k_proj<<<gp, 32, 0, stream>>>(qB[cur], WcatT, projT, M);
      long tea = (long)B_ * N * Hh;
      k_ea<<<(unsigned)((tea + 255) / 256), 256, 0, stream>>>(clipF_cur, wa1, eaBuf, N);
      long teb = (long)B_ * Hh * M;
      k_eb<<<(unsigned)((teb + 255) / 256), 256, 0, stream>>>(projT, av, ebBuf, M);
      k_smax_stats<<<(unsigned)((teb + 255) / 256), 256, 0, stream>>>(
          eaBuf, ebBuf, adj_q2c, colmax, colrcp, N, M);
      long tat = (long)B_ * Hh * N * M;
      k_att<<<(unsigned)((tat + 255) / 256), 256, 0, stream>>>(
          eaBuf, ebBuf, adj_q2c, colmax, colrcp, attW, N, M);
      dim3 gh(N / 32, B_ * Hh);
      k_hgemm<<<gh, 32, 0, stream>>>(attW, projT, aggB, N, M);
      dim3 gf(B_ * N / 32, NHIDd / 16);
      k_fusion<<<gf, 32, 0, stream>>>(clipB[cur], aggB, pT, clipF_cur, clipF_next, clipB[nxt]);
    }

    // ===== direction 2: aggregate clip into question (N=NQ, M=NC) =====
    {
      const float* W  = W2 + (size_t)l * Hh * Fdim * DHd;
      const float* av = a2 + (size_t)l * Hh * 2 * DHd;
      const float* fp = f2 + (size_t)l * 4 * Fdim * NHIDd;
      const int N = NQ, M = NC;
      k_prep_wcat<<<(NHIDd * Fdim + 255) / 256, 256, 0, stream>>>(W, WcatT);
      k_prep_wa1<<<(Hh * Fdim + 255) / 256, 256, 0, stream>>>(W, av, wa1);
      k_prep_pT<<<(4 * NHIDd * Fdim + 255) / 256, 256, 0, stream>>>(fp, pT);
      dim3 gp(B_ * M / 32, NHIDd / 64);
      k_proj<<<gp, 32, 0, stream>>>(clipB[cur], WcatT, projT, M);   // uses OLD clip
      long tea = (long)B_ * N * Hh;
      k_ea<<<(unsigned)((tea + 255) / 256), 256, 0, stream>>>(qF_cur, wa1, eaBuf, N);
      long teb = (long)B_ * Hh * M;
      k_eb<<<(unsigned)((teb + 255) / 256), 256, 0, stream>>>(projT, av, ebBuf, M);
      k_smax_stats<<<(unsigned)((teb + 255) / 256), 256, 0, stream>>>(
          eaBuf, ebBuf, adj_c2q, colmax, colrcp, N, M);
      long tat = (long)B_ * Hh * N * M;
      k_att<<<(unsigned)((tat + 255) / 256), 256, 0, stream>>>(
          eaBuf, ebBuf, adj_c2q, colmax, colrcp, attW, N, M);
      dim3 gh(N / 32, B_ * Hh);
      k_hgemm<<<gh, 32, 0, stream>>>(attW, projT, aggB, N, M);
      dim3 gf(B_ * N / 32, NHIDd / 16);
      k_fusion<<<gf, 32, 0, stream>>>(qB[cur], aggB, pT, qF_cur, qF_next, qB[nxt]);
    }

    clipF_cur = clipF_next;
    qF_cur    = qF_next;
    cur = nxt;
  }
}
